// RNNPredictor_38792144617752
// MI455X (gfx1250) — matrix operand order repaired and checked by reading, not compiled, not run on hardware
//
#include <hip/hip_runtime.h>
#include <math.h>

// Problem constants (from reference): B, T, V, E, U, OUT = 1024, 64, 30, 128, 512, 20
#define Bsz   1024
#define Tsz   64
#define Esz   128
#define Usz   512
#define Gsz   2048   // 4*U
#define OUTsz 20

typedef __attribute__((ext_vector_type(16))) __bf16        v16bf;
typedef __attribute__((ext_vector_type(8)))  float         v8f;
typedef __attribute__((ext_vector_type(4)))  unsigned int  u32x4;

union bfrag { v16bf v; u32x4 q[2]; };

__device__ __forceinline__ unsigned short f2bf(float f) {
  unsigned int u = __float_as_uint(f);
  u = (u + 0x7FFFu + ((u >> 16) & 1u)) >> 16;   // round-to-nearest-even
  return (unsigned short)u;
}
__device__ __forceinline__ float bf2f(unsigned short h) {
  return __uint_as_float(((unsigned int)h) << 16);
}
__device__ __forceinline__ float sigmoidf(float x) {
  return 1.0f / (1.0f + __expf(-x));
}

// ---- WMMA fragment loaders (bf16, 16x16x32) -------------------------------
// A (16x32, row-major, per ISA 16-bit A layout): lane L -> row L%16,
// halves 0..7 = K = 8*(L/16)+0..7 ; halves 8..15 = K = 16 + 8*(L/16)+0..7
__device__ __forceinline__ v16bf load_fragA(const unsigned short* A, int lda,
                                            int m0, int k0, int lane) {
  int r = lane & 15;
  int h = lane >> 4;
  const unsigned short* p = A + (size_t)(m0 + r) * lda + k0;
  bfrag f;
  f.q[0] = *(const u32x4*)(p + h * 8);
  f.q[1] = *(const u32x4*)(p + 16 + h * 8);
  return f.v;
}
// B (32x16) from transposed weights Bt[N,K] row-major: lane L -> col n0 + L%16,
// halves 0..7 = K = 8*(L/16)+0..7 ; halves 8..15 = K = 16 + 8*(L/16)+0..7
__device__ __forceinline__ v16bf load_fragB(const unsigned short* Bt, int ldb,
                                            int n0, int k0, int lane) {
  int c = lane & 15;
  int h = lane >> 4;
  const unsigned short* p = Bt + (size_t)(n0 + c) * ldb + k0;
  bfrag f;
  f.q[0] = *(const u32x4*)(p + h * 8);
  f.q[1] = *(const u32x4*)(p + 16 + h * 8);
  return f.v;
}

// ---- small helper kernels --------------------------------------------------
__global__ void embed_kernel(const int* __restrict__ x, const float* __restrict__ emb,
                             unsigned short* __restrict__ e, int total) {
  int idx = blockIdx.x * blockDim.x + threadIdx.x;
  if (idx >= total) return;
  int bt = idx / Esz;
  int k  = idx - bt * Esz;
  int tok = x[bt];
  e[idx] = f2bf(emb[tok * Esz + k]);
}

// Wt[n*K + k] = bf16(W[k*N + n])   (W is [K,N] row-major)
__global__ void cast_transpose_kernel(const float* __restrict__ W,
                                      unsigned short* __restrict__ Wt,
                                      int K, int N) {
  int idx = blockIdx.x * blockDim.x + threadIdx.x;
  if (idx >= K * N) return;
  int k = idx / N;
  int n = idx - k * N;
  Wt[(size_t)n * K + k] = f2bf(W[(size_t)k * N + n]);
}

// ---- fused LSTM step: z = x_t@W + h_{t-1}@U + b ; gates ; c,h update ------
// Each wave: 32 batch rows (2 M-subtiles) x 16 units x 4 gates.
// 8 WMMAs per 32-K chunk from 6 fragment loads (B shared by both A subtiles).
__global__ void __launch_bounds__(256)
lstm_step_kernel(const unsigned short* __restrict__ xt, int ldx, int Dx,
                 const unsigned short* __restrict__ Wt,        // [G, Dx]
                 const unsigned short* __restrict__ hprev, int ldh,
                 const unsigned short* __restrict__ Ut,        // [G, U]
                 const float* __restrict__ bias,               // [G]
                 float* __restrict__ c,                        // [B, U] in/out
                 unsigned short* __restrict__ hout, int ldho,
                 int first) {
  int lane = threadIdx.x & 31;
  int wave = threadIdx.x >> 5;
  int n0 = blockIdx.x * 16;                        // unit tile
  int m0 = (blockIdx.y * 8 + wave) * 32;           // batch tile (32 rows)

  v8f acc[2][4];
  {
    int n = lane & 15;
#pragma unroll
    for (int g = 0; g < 4; ++g) {
      float bv = bias[g * Usz + n0 + n];
#pragma unroll
      for (int v = 0; v < 8; ++v) { acc[0][g][v] = bv; acc[1][g][v] = bv; }
    }
  }

  // input contribution: x_t @ W
  for (int k0 = 0; k0 < Dx; k0 += 32) {
    v16bf a0 = load_fragA(xt, ldx, m0,      k0, lane);
    v16bf a1 = load_fragA(xt, ldx, m0 + 16, k0, lane);
#pragma unroll
    for (int g = 0; g < 4; ++g) {
      v16bf b = load_fragB(Wt, Dx, g * Usz + n0, k0, lane);
      acc[0][g] = __builtin_amdgcn_wmma_f32_16x16x32_bf16(
          false, a0, false, b, (short)0, acc[0][g], false, false);
      acc[1][g] = __builtin_amdgcn_wmma_f32_16x16x32_bf16(
          false, a1, false, b, (short)0, acc[1][g], false, false);
    }
  }
  // recurrent contribution: h_{t-1} @ U
  if (!first) {
    for (int k0 = 0; k0 < Usz; k0 += 32) {
      v16bf a0 = load_fragA(hprev, ldh, m0,      k0, lane);
      v16bf a1 = load_fragA(hprev, ldh, m0 + 16, k0, lane);
#pragma unroll
      for (int g = 0; g < 4; ++g) {
        v16bf b = load_fragB(Ut, Usz, g * Usz + n0, k0, lane);
        acc[0][g] = __builtin_amdgcn_wmma_f32_16x16x32_bf16(
            false, a0, false, b, (short)0, acc[0][g], false, false);
        acc[1][g] = __builtin_amdgcn_wmma_f32_16x16x32_bf16(
            false, a1, false, b, (short)0, acc[1][g], false, false);
      }
    }
  }

  // gates + state update (D layout: n = lane%16, m = mbase + 8*(lane/16) + v)
  int n = n0 + (lane & 15);
#pragma unroll
  for (int s = 0; s < 2; ++s) {
    int mb = m0 + 16 * s + ((lane >> 4) << 3);
#pragma unroll
    for (int v = 0; v < 8; ++v) {
      int m = mb + v;
      float iv = sigmoidf(acc[s][0][v]);
      float fv = sigmoidf(acc[s][1][v]);
      float gv = tanhf(acc[s][2][v]);
      float ov = sigmoidf(acc[s][3][v]);
      size_t ci = (size_t)m * Usz + n;
      float cn = first ? (iv * gv) : fmaf(fv, c[ci], iv * gv);
      c[ci] = cn;
      hout[(size_t)m * ldho + n] = f2bf(ov * tanhf(cn));
    }
  }
}

// ---- dense: C = relu(A @ Bt^T + bias), bf16 out (M=32 blocked) ------------
__global__ void __launch_bounds__(256)
gemm_bias_relu_kernel(const unsigned short* __restrict__ A, int lda,
                      const unsigned short* __restrict__ Bt, int ldb,
                      const float* __restrict__ bias,
                      unsigned short* __restrict__ C, int ldc, int K) {
  int lane = threadIdx.x & 31;
  int wave = threadIdx.x >> 5;
  int n0 = blockIdx.x * 16;
  int m0 = (blockIdx.y * 8 + wave) * 32;

  v8f acc[2];
  float bv = bias[n0 + (lane & 15)];
#pragma unroll
  for (int v = 0; v < 8; ++v) { acc[0][v] = bv; acc[1][v] = bv; }

  for (int k0 = 0; k0 < K; k0 += 32) {
    v16bf a0 = load_fragA(A, lda, m0,      k0, lane);
    v16bf a1 = load_fragA(A, lda, m0 + 16, k0, lane);
    v16bf b  = load_fragB(Bt, ldb, n0, k0, lane);
    acc[0] = __builtin_amdgcn_wmma_f32_16x16x32_bf16(
        false, a0, false, b, (short)0, acc[0], false, false);
    acc[1] = __builtin_amdgcn_wmma_f32_16x16x32_bf16(
        false, a1, false, b, (short)0, acc[1], false, false);
  }

  int n = n0 + (lane & 15);
#pragma unroll
  for (int s = 0; s < 2; ++s) {
    int mb = m0 + 16 * s + ((lane >> 4) << 3);
#pragma unroll
    for (int v = 0; v < 8; ++v) {
      float x = acc[s][v];
      x = x > 0.0f ? x : 0.0f;   // relu
      C[(size_t)(mb + v) * ldc + n] = f2bf(x);
    }
  }
}

// ---- final projection: out[b,o] = d[b,:] @ Wo[:,o] + bo[o]  (N=20) --------
__global__ void out_proj_kernel(const unsigned short* __restrict__ d,
                                const float* __restrict__ Wo,
                                const float* __restrict__ bo,
                                float* __restrict__ out) {
  int idx = blockIdx.x * blockDim.x + threadIdx.x;
  if (idx >= Bsz * OUTsz) return;
  int b = idx / OUTsz;
  int o = idx - b * OUTsz;
  float s = bo[o];
  const unsigned short* dr = d + (size_t)b * (2 * Usz);
  for (int k = 0; k < 2 * Usz; ++k)
    s = fmaf(bf2f(dr[k]), Wo[(size_t)k * OUTsz + o], s);
  out[idx] = s;
}

// ---------------------------------------------------------------------------
extern "C" void kernel_launch(void* const* d_in, const int* in_sizes, int n_in,
                              void* d_out, int out_size, void* d_ws, size_t ws_size,
                              hipStream_t stream) {
  (void)in_sizes; (void)n_in; (void)out_size; (void)ws_size;
  const int*   x   = (const int*)  d_in[0];
  const float* emb = (const float*)d_in[1];
  const float* W1f = (const float*)d_in[2];
  const float* U1f = (const float*)d_in[3];
  const float* b1f = (const float*)d_in[4];
  const float* W1b = (const float*)d_in[5];
  const float* U1b = (const float*)d_in[6];
  const float* b1b = (const float*)d_in[7];
  const float* W2f = (const float*)d_in[8];
  const float* U2f = (const float*)d_in[9];
  const float* b2f = (const float*)d_in[10];
  const float* W2b = (const float*)d_in[11];
  const float* U2b = (const float*)d_in[12];
  const float* b2b = (const float*)d_in[13];
  const float* Wd  = (const float*)d_in[14];
  const float* bd  = (const float*)d_in[15];
  const float* Wo  = (const float*)d_in[16];
  const float* bo  = (const float*)d_in[17];
  float* out = (float*)d_out;

  // ---- workspace carve-out (bf16 buffers as ushort) ----
  char* ws = (char*)d_ws;
  size_t off = 0;
  auto carve = [&](size_t bytes) -> char* {
    char* p = ws + off;
    off += (bytes + 255) & ~(size_t)255;
    return p;
  };
  unsigned short* e    = (unsigned short*)carve((size_t)Bsz * Tsz * Esz * 2);
  unsigned short* h1   = (unsigned short*)carve((size_t)Bsz * Tsz * 2 * Usz * 2);
  unsigned short* W1fT = (unsigned short*)carve((size_t)Gsz * Esz * 2);
  unsigned short* W1bT = (unsigned short*)carve((size_t)Gsz * Esz * 2);
  unsigned short* U1fT = (unsigned short*)carve((size_t)Gsz * Usz * 2);
  unsigned short* U1bT = (unsigned short*)carve((size_t)Gsz * Usz * 2);
  unsigned short* W2fT = (unsigned short*)carve((size_t)Gsz * 2 * Usz * 2);
  unsigned short* W2bT = (unsigned short*)carve((size_t)Gsz * 2 * Usz * 2);
  unsigned short* U2fT = (unsigned short*)carve((size_t)Gsz * Usz * 2);
  unsigned short* U2bT = (unsigned short*)carve((size_t)Gsz * Usz * 2);
  unsigned short* WdT  = (unsigned short*)carve((size_t)2 * Usz * 2 * Usz * 2);
  float*          cA   = (float*)        carve((size_t)Bsz * Usz * 4);
  unsigned short* hq0  = (unsigned short*)carve((size_t)Bsz * Usz * 2);
  unsigned short* hq1  = (unsigned short*)carve((size_t)Bsz * Usz * 2);
  unsigned short* h2   = (unsigned short*)carve((size_t)Bsz * 2 * Usz * 2);
  unsigned short* dbuf = (unsigned short*)carve((size_t)Bsz * 2 * Usz * 2);

  // ---- weight prep (cast fp32 -> bf16, transpose to [N,K]) ----
  auto ct = [&](const float* W, unsigned short* Wt, int K, int N) {
    int total = K * N;
    cast_transpose_kernel<<<(total + 255) / 256, 256, 0, stream>>>(W, Wt, K, N);
  };
  ct(W1f, W1fT, Esz, Gsz);
  ct(W1b, W1bT, Esz, Gsz);
  ct(U1f, U1fT, Usz, Gsz);
  ct(U1b, U1bT, Usz, Gsz);
  ct(W2f, W2fT, 2 * Usz, Gsz);
  ct(W2b, W2bT, 2 * Usz, Gsz);
  ct(U2f, U2fT, Usz, Gsz);
  ct(U2b, U2bT, Usz, Gsz);
  ct(Wd,  WdT,  2 * Usz, 2 * Usz);

  // ---- embedding gather to bf16 ----
  {
    int total = Bsz * Tsz * Esz;
    embed_kernel<<<(total + 255) / 256, 256, 0, stream>>>(x, emb, e, total);
  }

  dim3 sgrid(Usz / 16, Bsz / (32 * 8));   // (32, 4)
  dim3 sblk(256);
  auto step = [&](const unsigned short* xt, int ldx, int Dx,
                  const unsigned short* Wt,
                  const unsigned short* hprev, int ldh,
                  const unsigned short* Ut, const float* bias, float* c,
                  unsigned short* hout, int ldho, int first) {
    lstm_step_kernel<<<sgrid, sblk, 0, stream>>>(xt, ldx, Dx, Wt, hprev, ldh,
                                                 Ut, bias, c, hout, ldho, first);
  };

  const int ld_h1 = Tsz * 2 * Usz;   // batch-row stride of h1[B,T,2U]
  const int ld_e  = Tsz * Esz;

  // ---- layer 1 forward (writes h1[:, t, 0:U]) ----
  for (int t = 0; t < Tsz; ++t) {
    step(e + (size_t)t * Esz, ld_e, Esz, W1fT,
         t ? h1 + (size_t)(t - 1) * 2 * Usz : (const unsigned short*)nullptr, ld_h1,
         U1fT, b1f, cA,
         h1 + (size_t)t * 2 * Usz, ld_h1, t == 0);
  }
  // ---- layer 1 backward (writes h1[:, t, U:2U]) ----
  for (int s = 0; s < Tsz; ++s) {
    int t = Tsz - 1 - s;
    step(e + (size_t)t * Esz, ld_e, Esz, W1bT,
         s ? h1 + (size_t)(t + 1) * 2 * Usz + Usz : (const unsigned short*)nullptr, ld_h1,
         U1bT, b1b, cA,
         h1 + (size_t)t * 2 * Usz + Usz, ld_h1, s == 0);
  }
  // ---- layer 2 forward (keep last h only) ----
  for (int t = 0; t < Tsz; ++t) {
    const unsigned short* hp = t ? (((t - 1) & 1) ? hq1 : hq0)
                                 : (const unsigned short*)nullptr;
    unsigned short* ho;
    int ldho;
    if (t == Tsz - 1) { ho = h2;                 ldho = 2 * Usz; }
    else              { ho = (t & 1) ? hq1 : hq0; ldho = Usz;    }
    step(h1 + (size_t)t * 2 * Usz, ld_h1, 2 * Usz, W2fT, hp, Usz,
         U2fT, b2f, cA, ho, ldho, t == 0);
  }
  // ---- layer 2 backward (keep last h only; consumes t = T-1 .. 0) ----
  for (int s = 0; s < Tsz; ++s) {
    int t = Tsz - 1 - s;
    const unsigned short* hp = s ? (((s - 1) & 1) ? hq1 : hq0)
                                 : (const unsigned short*)nullptr;
    unsigned short* ho;
    int ldho;
    if (s == Tsz - 1) { ho = h2 + Usz;            ldho = 2 * Usz; }
    else              { ho = (s & 1) ? hq1 : hq0; ldho = Usz;    }
    step(h1 + (size_t)t * 2 * Usz, ld_h1, 2 * Usz, W2bT, hp, Usz,
         U2bT, b2b, cA, ho, ldho, s == 0);
  }

  // ---- dense + relu:  d = relu(h2 @ Wd + bd)  (WMMA, M=32 blocked) ----
  gemm_bias_relu_kernel<<<dim3((2 * Usz) / 16, Bsz / (32 * 8)), 256, 0, stream>>>(
      h2, 2 * Usz, WdT, 2 * Usz, bd, dbuf, 2 * Usz, 2 * Usz);

  // ---- output projection (N=20, trivial fp32 kernel) ----
  out_proj_kernel<<<(Bsz * OUTsz + 255) / 256, 256, 0, stream>>>(dbuf, Wo, bo, out);
}
